// TrajectoryPrediction_58832462020776
// MI455X (gfx1250) — matrix operand order repaired and checked by reading, not compiled, not run on hardware
//
#include <hip/hip_runtime.h>
#include <hip/hip_bf16.h>
#include <math.h>

typedef __attribute__((ext_vector_type(16))) _Float16 v16h;
typedef __attribute__((ext_vector_type(8)))  _Float16 v8h;
typedef __attribute__((ext_vector_type(8)))  float    v8f;

#define T_OBS 8
#define NPED  2048
#define NGRP  64
#define GSZ   32
#define HDIM  128
#define DIN   64
#define DDD   64
#define NHEAD 4

__device__ __forceinline__ float sigf(float x) { return 1.0f / (1.0f + __expf(-x)); }

// ---- WMMA fragment builders (CDNA5 16x16x32 f16, wave32 layouts, ISA 7.12.2) ----

// A (16x32, MxK) from f16 row (global or LDS); halves 0-7 at kbase, 8-15 at kbase+16.
// kbase = ks*32 + (lane>>4)*8  -> 16B-aligned vector loads.
__device__ __forceinline__ v16h load_a_h16(const _Float16* row, int kbase) {
  v8h lo = *(const v8h*)(row + kbase);
  v8h hi = *(const v8h*)(row + kbase + 16);
  v16h a;
#pragma unroll
  for (int h = 0; h < 8; ++h) { a[h] = lo[h]; a[8 + h] = hi[h]; }
  return a;
}

// B (32x16, KxN): halves 0-7 at kbase, 8-15 at kbase+16; kbase = ks*32 + (lane>>4)*8.
__device__ __forceinline__ v16h load_b_h16(const _Float16* row, int kbase) {
  v8h lo = *(const v8h*)(row + kbase);
  v8h hi = *(const v8h*)(row + kbase + 16);
  v16h b;
#pragma unroll
  for (int h = 0; h < 8; ++h) { b[h] = lo[h]; b[8 + h] = hi[h]; }
  return b;
}

__device__ __forceinline__ v8f wmma32(v16h a, v16h b, v8f c) {
  return __builtin_amdgcn_wmma_f32_16x16x32_f16(false, a, false, b, (short)0, c, false, false);
}

// ======================= Kernel 0: weight prep (one-time fp32->fp16 + folds) ===========
// Wih16 = f16(W_ih)               [512][64]
// Whh16 = f16(W_hh)               [512][128]
// wT16[h][o][k] = f16(gat_w[h][k][o])   (transposed so Phase-E B fragments are contiguous)
// wa[0:512]  = wa1[h][f] = sum_o w[h][f][o]*a[o]
// wa[512:1024]= wa2[h][f] = sum_o w[h][f][o]*a[128+o]
__global__ __launch_bounds__(256) void prep_kernel(const float* __restrict__ W_ih,
                                                   const float* __restrict__ W_hh,
                                                   const float* __restrict__ gat_w,
                                                   const float* __restrict__ gat_a,
                                                   _Float16* __restrict__ Wih16,
                                                   _Float16* __restrict__ Whh16,
                                                   _Float16* __restrict__ wT16,
                                                   float* __restrict__ wa) {
  int idx = blockIdx.x * 256 + threadIdx.x;           // up to 65536
  if (idx < 512 * 64) Wih16[idx] = (_Float16)W_ih[idx];
  if (idx < 512 * 128) Whh16[idx] = (_Float16)W_hh[idx];
  if (idx < NHEAD * HDIM * HDIM) {
    int h = idx >> 14, rem = idx & 16383, o = rem >> 7, k = rem & 127;
    wT16[idx] = (_Float16)gat_w[(size_t)h * HDIM * HDIM + k * HDIM + o];
  }
  if (idx < NHEAD * HDIM) {
    int h = idx >> 7, f = idx & 127;
    const float* wrow = gat_w + (size_t)h * HDIM * HDIM + (size_t)f * HDIM;
    float s1 = 0.f, s2 = 0.f;
#pragma unroll 4
    for (int o = 0; o < HDIM; ++o) { float w = wrow[o]; s1 += w * gat_a[o]; s2 += w * gat_a[HDIM + o]; }
    wa[idx] = s1;
    wa[NHEAD * HDIM + idx] = s2;
  }
}

// ======================= Kernel 1: point embedding (writes f16, WMMA-ready) ============
__global__ __launch_bounds__(256) void embed_kernel(const float* __restrict__ obs,
                                                    const float* __restrict__ W_emb,
                                                    const float* __restrict__ b_emb,
                                                    _Float16* __restrict__ xout) {
  int idx = blockIdx.x * 256 + threadIdx.x;           // [0, T*N*DIN)
  if (idx >= T_OBS * NPED * DIN) return;
  int d = idx & 63;
  int row = idx >> 6;                                 // t*N + p
  float px = obs[row * 2 + 0];
  float py = obs[row * 2 + 1];
  float v = px * W_emb[d] + py * W_emb[64 + d] + b_emb[d];
  xout[idx] = (_Float16)fmaxf(v, 0.0f);
}

// ======================= Kernel 2: LSTM (8 steps), WMMA gates ==========================
// One block = 16 pedestrians; 8 wave32s each own 4 of the 32 gate N-tiles (512 gate cols).
__global__ __launch_bounds__(256) void lstm_kernel(const _Float16* __restrict__ xh,
                                                   const float* __restrict__ h0,
                                                   const float* __restrict__ c0,
                                                   const _Float16* __restrict__ Wih16,
                                                   const _Float16* __restrict__ Whh16,
                                                   const float* __restrict__ b_ih,
                                                   const float* __restrict__ b_hh,
                                                   float* __restrict__ ah_out) {
  __shared__ float cbuf[16][HDIM];
  __shared__ float hbuf[16][HDIM];
  __shared__ __align__(32) _Float16 h16[16][HDIM];    // f16 mirror for A fragments
  __shared__ float gbuf[16][4 * HDIM];

  const int ped0 = blockIdx.x * 16;
  const int tid  = threadIdx.x;
  const int lane = tid & 31;
  const int wv   = tid >> 5;

  for (int idx = tid; idx < 16 * HDIM; idx += 256) {
    int m = idx >> 7, f = idx & 127;
    float h = h0[(ped0 + m) * HDIM + f];
    hbuf[m][f] = h;
    h16[m][f] = (_Float16)h;
    cbuf[m][f] = c0[(ped0 + m) * HDIM + f];
  }
  __syncthreads();

  const int mrow = lane & 15;
  const int kbA  = (lane >> 4) * 8;       // A-fragment K base within 32-chunk
  const int kbB  = (lane >> 4) * 8;       // B-fragment K base within 32-chunk

  for (int t = 0; t < T_OBS; ++t) {
    const _Float16* xrow = xh + (size_t)(t * NPED + ped0 + mrow) * DIN;

    for (int tt = 0; tt < 4; ++tt) {
      const int jt = wv * 4 + tt;                  // gate N-tile 0..31
      const int jn = jt * 16 + (lane & 15);        // this lane's gate column
      const _Float16* wihRow = Wih16 + (size_t)jn * DIN;
      const _Float16* whhRow = Whh16 + (size_t)jn * HDIM;
      __builtin_prefetch(whhRow, 0, 0);

      const float bias = b_ih[jn] + b_hh[jn];
      v8f acc;
#pragma unroll
      for (int v = 0; v < 8; ++v) acc[v] = bias;

      // x @ W_ih^T  (K = 64, 2 WMMA)
#pragma unroll
      for (int ks = 0; ks < 2; ++ks) {
        v16h a = load_a_h16(xrow, ks * 32 + kbA);
        v16h b = load_b_h16(wihRow, ks * 32 + kbB);
        acc = wmma32(a, b, acc);
      }
      // h @ W_hh^T  (K = 128, 4 WMMA)
#pragma unroll
      for (int ks = 0; ks < 4; ++ks) {
        v16h a = load_a_h16(&h16[mrow][0], ks * 32 + kbA);
        v16h b = load_b_h16(whhRow, ks * 32 + kbB);
        acc = wmma32(a, b, acc);
      }
      // store C tile: M = v + (lane>>4)*8, N = jn
#pragma unroll
      for (int v = 0; v < 8; ++v) gbuf[v + (lane >> 4) * 8][jn] = acc[v];
    }
    __syncthreads();

    // elementwise gate update
    for (int idx = tid; idx < 16 * HDIM; idx += 256) {
      int m = idx >> 7, f = idx & 127;
      float ig = gbuf[m][f];
      float fg = gbuf[m][HDIM + f];
      float gg = gbuf[m][2 * HDIM + f];
      float og = gbuf[m][3 * HDIM + f];
      float c = sigf(fg) * cbuf[m][f] + sigf(ig) * tanhf(gg);
      float h = sigf(og) * tanhf(c);
      cbuf[m][f] = c;
      hbuf[m][f] = h;
      h16[m][f] = (_Float16)h;
    }
    __syncthreads();
  }

  for (int idx = tid; idx < 16 * HDIM; idx += 256) {
    int m = idx >> 7, f = idx & 127;
    ah_out[(ped0 + m) * HDIM + f] = hbuf[m][f];
  }
}

// ======================= Kernel 3: GAT (fused online softmax + WMMA) ===================
// One block per group (n=32). scores[h,i,k] = ah[i]·wa1[h] + h_other[i,k]·wa2[h];
// ctx[h,i] = softmax-weighted sum of h_other rows; out[i] = relu(mean_h ctx[h,i]@w[h]) + bias.
__global__ __launch_bounds__(256) void gat_kernel(const float* __restrict__ ah_g,
                                                  const float* __restrict__ gh_g,
                                                  const float* __restrict__ gl_g,
                                                  const float* __restrict__ ac_g,
                                                  const float* __restrict__ Wd_g,
                                                  const float* __restrict__ bd_g,
                                                  const float* __restrict__ Wg_g,
                                                  const float* __restrict__ bg_g,
                                                  const _Float16* __restrict__ wT16,
                                                  const float* __restrict__ wa_g,
                                                  const float* __restrict__ gbias_g,
                                                  float* __restrict__ out) {
  __shared__ _Float16 tah[GSZ][HDIM];       // tanh(action_hidden), f16
  __shared__ float    wa1[NHEAD][HDIM];     // w[h] @ a[0:128]
  __shared__ float    wa2s[NHEAD][HDIM];    // w[h] @ a[128:256]
  __shared__ float    si1[NHEAD][GSZ];      // ah[i]·wa1[h]
  __shared__ float    m1s[8][DDD];          // per-wave MLP hidden staging
  __shared__ _Float16 Wd[8][DDD];
  __shared__ _Float16 Wg[DDD][HDIM];
  __shared__ float    bd[DDD], bg[HDIM];
  __shared__ float    acs[GSZ][2], gls[GSZ][2];
  __shared__ __align__(32) _Float16 ctx[NHEAD][GSZ][HDIM];

  const int grp  = blockIdx.x;
  const int tid  = threadIdx.x;
  const int lane = tid & 31;
  const int wv   = tid >> 5;

  const float* ah = ah_g + (size_t)grp * GSZ * HDIM;
  const float* gh = gh_g + (size_t)grp * GSZ * HDIM;

  // ---- Phase A: stage inputs ----
  for (int idx = tid; idx < GSZ * HDIM; idx += 256) {
    int i = idx >> 7, f = idx & 127;
    tah[i][f] = (_Float16)tanhf(ah[i * HDIM + f]);
  }
  for (int idx = tid; idx < 8 * DDD; idx += 256) Wd[idx >> 6][idx & 63] = (_Float16)Wd_g[idx];
  for (int idx = tid; idx < DDD * HDIM; idx += 256) Wg[idx >> 7][idx & 127] = (_Float16)Wg_g[idx];
  for (int idx = tid; idx < NHEAD * HDIM; idx += 256) {
    wa1[idx >> 7][idx & 127]  = wa_g[idx];
    wa2s[idx >> 7][idx & 127] = wa_g[NHEAD * HDIM + idx];
  }
  if (tid < DDD) bd[tid] = bd_g[tid];
  if (tid < HDIM) bg[tid] = bg_g[tid];
  if (tid < GSZ) {
    acs[tid][0] = ac_g[(grp * GSZ + tid) * 2 + 0];
    acs[tid][1] = ac_g[(grp * GSZ + tid) * 2 + 1];
    gls[tid][0] = gl_g[(grp * GSZ + tid) * 2 + 0];
    gls[tid][1] = gl_g[(grp * GSZ + tid) * 2 + 1];
  }
  __syncthreads();
  if (tid < NHEAD * GSZ) {
    int h = tid >> 5, i = tid & 31;
    float s = 0.f;
#pragma unroll 4
    for (int f = 0; f < HDIM; ++f) s += ah[i * HDIM + f] * wa1[h][f];
    si1[h][i] = s;
  }
  __syncthreads();

  // per-lane slice of wa2 (o = lane*4 + c)
  float w2l[NHEAD][4];
#pragma unroll
  for (int h = 0; h < NHEAD; ++h)
#pragma unroll
    for (int c = 0; c < 4; ++c) w2l[h][c] = wa2s[h][lane * 4 + c];

  // ---- Phase B+D fused: online softmax over k=0..32 with context accumulation ----
  for (int ii = 0; ii < 4; ++ii) {
    const int i = wv * 4 + ii;
    float mrun[NHEAD], srun[NHEAD], acc4[NHEAD][4];
#pragma unroll
    for (int h = 0; h < NHEAD; ++h) {
      mrun[h] = -__builtin_inff();
      srun[h] = 0.f;
#pragma unroll
      for (int c = 0; c < 4; ++c) acc4[h][c] = 0.f;
    }

    // k = 0: h_other = ah[i]
    {
      float vec[4];
#pragma unroll
      for (int c = 0; c < 4; ++c) vec[c] = ah[i * HDIM + lane * 4 + c];
#pragma unroll
      for (int h = 0; h < NHEAD; ++h) {
        float p = vec[0] * w2l[h][0] + vec[1] * w2l[h][1] + vec[2] * w2l[h][2] + vec[3] * w2l[h][3];
        for (int off = 16; off >= 1; off >>= 1) p += __shfl_xor(p, off, 32);
        float sc = si1[h][i] + p;
        sc = (sc >= 0.f) ? sc : 0.2f * sc;          // leaky_relu(0.2)
        float mn = fmaxf(mrun[h], sc);
        float scale = __expf(mrun[h] - mn);
        float e = __expf(sc - mn);
        srun[h] = srun[h] * scale + e;
#pragma unroll
        for (int c = 0; c < 4; ++c) acc4[h][c] = acc4[h][c] * scale + e * vec[c];
        mrun[h] = mn;
      }
    }

    const float d0 = acs[i][0], d1 = acs[i][1], d2 = gls[i][0], d3 = gls[i][1];

    for (int j = 0; j < GSZ; ++j) {
      const float d4 = acs[j][0], d5 = acs[j][1], d6 = gls[j][0], d7 = gls[j][1];
      // MLP hidden: each lane produces 2 of 64
#pragma unroll
      for (int qq = 0; qq < 2; ++qq) {
        int q = lane * 2 + qq;
        float s = bd[q];
        s += d0 * (float)Wd[0][q] + d1 * (float)Wd[1][q] + d2 * (float)Wd[2][q] + d3 * (float)Wd[3][q];
        s += d4 * (float)Wd[4][q] + d5 * (float)Wd[5][q] + d6 * (float)Wd[6][q] + d7 * (float)Wd[7][q];
        m1s[wv][q] = fmaxf(s, 0.f);
      }
      __syncthreads();   // uniform trip counts across all waves

      float vec[4];
#pragma unroll
      for (int c = 0; c < 4; ++c) {
        int o = lane * 4 + c;
        float s = bg[o];
#pragma unroll 8
        for (int q = 0; q < DDD; ++q) s += m1s[wv][q] * (float)Wg[q][o];
        float g = sigf(s);
        vec[c] = g * (float)tah[j][o];              // gate_action
      }
      if (j == i) {                                  // diagonal -> goal hidden state
#pragma unroll
        for (int c = 0; c < 4; ++c) vec[c] = gh[i * HDIM + lane * 4 + c];
      }
      __syncthreads();   // WAR on m1s before next iteration

#pragma unroll
      for (int h = 0; h < NHEAD; ++h) {
        float p = vec[0] * w2l[h][0] + vec[1] * w2l[h][1] + vec[2] * w2l[h][2] + vec[3] * w2l[h][3];
        for (int off = 16; off >= 1; off >>= 1) p += __shfl_xor(p, off, 32);
        float sc = si1[h][i] + p;
        sc = (sc >= 0.f) ? sc : 0.2f * sc;
        float mn = fmaxf(mrun[h], sc);
        float scale = __expf(mrun[h] - mn);
        float e = __expf(sc - mn);
        srun[h] = srun[h] * scale + e;
#pragma unroll
        for (int c = 0; c < 4; ++c) acc4[h][c] = acc4[h][c] * scale + e * vec[c];
        mrun[h] = mn;
      }
    }

    // finalize context rows (f16 for WMMA A operand)
#pragma unroll
    for (int h = 0; h < NHEAD; ++h) {
      float inv = 1.0f / srun[h];
#pragma unroll
      for (int c = 0; c < 4; ++c) ctx[h][i][lane * 4 + c] = (_Float16)(acc4[h][c] * inv);
    }
  }
  __syncthreads();

  // ---- Phase E: out[i,:] = relu( (1/4) * sum_h ctx[h] @ w[h] ) + bias  (WMMA) ----
  const int kbA = (lane >> 4) * 8;
  const int kbB = (lane >> 4) * 8;
  for (int tt = 0; tt < 2; ++tt) {
    const int tile = wv * 2 + tt;     // 16 tiles: 2 M-tiles x 8 N-tiles
    const int mt = tile >> 3;
    const int nt = tile & 7;
    const int o  = nt * 16 + (lane & 15);
    v8f acc = {0.f, 0.f, 0.f, 0.f, 0.f, 0.f, 0.f, 0.f};
#pragma unroll
    for (int h = 0; h < NHEAD; ++h) {
      const _Float16* crow = &ctx[h][mt * 16 + (lane & 15)][0];
      const _Float16* wrow = wT16 + ((size_t)h * HDIM + o) * HDIM;   // wT[h][o][*]
#pragma unroll
      for (int ks = 0; ks < 4; ++ks) {
        v16h a = load_a_h16(crow, ks * 32 + kbA);
        v16h b = load_b_h16(wrow, ks * 32 + kbB);
        acc = wmma32(a, b, acc);
      }
    }
    float bias = gbias_g[o];
#pragma unroll
    for (int v = 0; v < 8; ++v) {
      int m = v + (lane >> 4) * 8;
      float val = fmaxf(acc[v] * 0.25f, 0.f) + bias;
      out[(size_t)(grp * GSZ + mt * 16 + m) * HDIM + o] = val;
    }
  }
}

// ======================= host launcher =======================
extern "C" void kernel_launch(void* const* d_in, const int* in_sizes, int n_in,
                              void* d_out, int out_size, void* d_ws, size_t ws_size,
                              hipStream_t stream) {
  const float* obs    = (const float*)d_in[0];
  const float* ghs    = (const float*)d_in[1];
  const float* goal   = (const float*)d_in[2];
  const float* action = (const float*)d_in[3];
  const float* h0     = (const float*)d_in[4];
  const float* c0     = (const float*)d_in[5];
  const float* W_emb  = (const float*)d_in[6];
  const float* b_emb  = (const float*)d_in[7];
  const float* W_ih   = (const float*)d_in[8];
  const float* W_hh   = (const float*)d_in[9];
  const float* b_ih   = (const float*)d_in[10];
  const float* b_hh   = (const float*)d_in[11];
  const float* W_dist = (const float*)d_in[12];
  const float* b_dist = (const float*)d_in[13];
  const float* W_gate = (const float*)d_in[14];
  const float* b_gate = (const float*)d_in[15];
  const float* gat_w  = (const float*)d_in[16];
  const float* gat_a  = (const float*)d_in[17];
  const float* gat_b  = (const float*)d_in[18];
  // d_in[19] = seq_start_end (int64) — groups are uniform (64 x 32), derived statically.

  // Workspace carve (all segment sizes keep 32B alignment):
  _Float16* xh    = (_Float16*)d_ws;                         // 8*2048*64 halfs (2 MB)
  _Float16* Wih16 = xh + (size_t)T_OBS * NPED * DIN;         // 512*64
  _Float16* Whh16 = Wih16 + 512 * DIN;                       // 512*128
  _Float16* wT16  = Whh16 + 512 * HDIM;                      // 4*128*128
  float*    wa_ws = (float*)(wT16 + NHEAD * HDIM * HDIM);    // 1024 f32
  float*    ahbuf = wa_ws + 2 * NHEAD * HDIM;                // 2048*128 f32 (1 MB)

  prep_kernel<<<(NHEAD * HDIM * HDIM + 255) / 256, 256, 0, stream>>>(
      W_ih, W_hh, gat_w, gat_a, Wih16, Whh16, wT16, wa_ws);
  embed_kernel<<<(T_OBS * NPED * DIN + 255) / 256, 256, 0, stream>>>(obs, W_emb, b_emb, xh);
  lstm_kernel<<<NPED / 16, 256, 0, stream>>>(xh, h0, c0, Wih16, Whh16, b_ih, b_hh, ahbuf);
  gat_kernel<<<NGRP, 256, 0, stream>>>(ahbuf, ghs, goal, action, W_dist, b_dist,
                                       W_gate, b_gate, wT16, wa_ws, gat_b, (float*)d_out);
}
